// Attention_76450417868987
// MI455X (gfx1250) — compile-verified, matrix operand order checked by reading, not run on hardware
//
#include <hip/hip_runtime.h>
#include <hip/hip_bf16.h>
#include <math.h>

// ---------------------------------------------------------------------------
// MI455X (gfx1250) implementation.
//  - GEMMs via v_wmma_f32_16x16x32_bf16 (f32 accumulate), one 16x16 tile/wave.
//  - Attention streaming fused per batch row; all streams use b128 loads
//    (memory-roofline bound: ~620MB of feature reads => ~27us at 23.3 TB/s).
// ---------------------------------------------------------------------------

typedef __attribute__((ext_vector_type(16))) __bf16 v16bf;
typedef __attribute__((ext_vector_type(8)))  float  v8f;

#define RNN 512
#define BATCH 512

// ---------------------------------------------------------------------------
// Generic WMMA GEMM:  D = act( A1@W1 [+ A2@W2] + bias1 + bias2 + addin )
// A row-major [M,K], W row-major [K,N].  M,N multiples of 16; N multiple of 64
// per grid config.  One 16x16 output tile per wave, 4 waves per block.
// ---------------------------------------------------------------------------
__global__ __launch_bounds__(128)
void wmma_gemm_kernel(const float* __restrict__ A1, int lda1,
                      const float* __restrict__ W1, int ldw1, int K1,
                      const float* __restrict__ A2, int lda2,
                      const float* __restrict__ W2, int ldw2, int K2,
                      const float* __restrict__ bias1,
                      const float* __restrict__ bias2,
                      const float* __restrict__ addin, int ldadd,
                      float* __restrict__ D, int ldd,
                      int act)
{
    const int lane = threadIdx.x & 31;
    const int wave = threadIdx.x >> 5;
    const int hl   = lane >> 4;        // half-wave (0/1)
    const int ml   = lane & 15;        // A row / B col within tile
    const int m0   = blockIdx.y * 16;
    const int n0   = (blockIdx.x * 4 + wave) * 16;

    v8f acc = {0.f, 0.f, 0.f, 0.f, 0.f, 0.f, 0.f, 0.f};

    for (int kk = 0; kk < K1; kk += 32) {
        v16bf af, bf;
        #pragma unroll
        for (int j = 0; j < 16; ++j) {
            const int v  = j >> 1, lo = j & 1;
            const int kl = (v < 4 ? 2 * v + lo : 16 + 2 * (v - 4) + lo) + hl * 8;
            af[j] = (__bf16)A1[(size_t)(m0 + ml) * lda1 + kk + kl];
            bf[j] = (__bf16)W1[(size_t)(kk + kl) * ldw1 + n0 + ml];
        }
        acc = __builtin_amdgcn_wmma_f32_16x16x32_bf16(
                  false, af, false, bf, (short)0, acc, false, false);
    }
    for (int kk = 0; kk < K2; kk += 32) {
        v16bf af, bf;
        #pragma unroll
        for (int j = 0; j < 16; ++j) {
            const int v  = j >> 1, lo = j & 1;
            const int kl = (v < 4 ? 2 * v + lo : 16 + 2 * (v - 4) + lo) + hl * 8;
            af[j] = (__bf16)A2[(size_t)(m0 + ml) * lda2 + kk + kl];
            bf[j] = (__bf16)W2[(size_t)(kk + kl) * ldw2 + n0 + ml];
        }
        acc = __builtin_amdgcn_wmma_f32_16x16x32_bf16(
                  false, af, false, bf, (short)0, acc, false, false);
    }

    float badd = 0.f;
    if (bias1) badd += bias1[n0 + ml];
    if (bias2) badd += bias2[n0 + ml];

    #pragma unroll
    for (int r = 0; r < 8; ++r) {
        const int row = m0 + hl * 8 + r;       // C/D layout: VGPR r -> M = r + half*8
        float v = acc[r] + badd;
        if (addin) v += addin[(size_t)row * ldadd + n0 + ml];
        if (act)   v = tanhf(v);
        D[(size_t)row * ldd + n0 + ml] = v;
    }
}

// ---------------------------------------------------------------------------
// Fused additive attention for one batch row per block:
//   scores[n] = sum_a tanh(p[b,n,a] + hp[b,a]) * aw[a]  (+ab)
//   w = softmax(scores);  out[b,:] = sum_n w[n] * feats[b,n,:]
// 256 threads = 8 waves.  Every stream uses float4 (b128) loads.
// ---------------------------------------------------------------------------
__global__ __launch_bounds__(256)
void attn_kernel(const float* __restrict__ hp, int ldhp,
                 const float* __restrict__ p_feats,
                 const float* __restrict__ feats,
                 const float* __restrict__ aw,
                 const float* __restrict__ ab,
                 float* __restrict__ outp, int ldout,
                 int N)
{
    __shared__ __align__(16) float hp_s[RNN];
    __shared__ __align__(16) float aw_s[RNN];
    __shared__ __align__(16) float sc[256];
    __shared__ __align__(16) float partial[RNN];

    const int b    = blockIdx.x;
    const int tid  = threadIdx.x;
    const int lane = tid & 31;
    const int wave = tid >> 5;

    // stage hp row + aw into LDS (float2 per thread)
    ((float2*)hp_s)[tid] = ((const float2*)(hp + (size_t)b * ldhp))[tid];
    ((float2*)aw_s)[tid] = ((const float2*)aw)[tid];
    __syncthreads();

    const size_t pb = (size_t)b * N * RNN;

    // ---- scores: one n per wave at a time; lanes cover dims as float4
    const float4* hp4 = (const float4*)hp_s;
    const float4* aw4 = (const float4*)aw_s;
    for (int n = wave; n < N; n += 8) {
        const float4* prow = (const float4*)(p_feats + pb + (size_t)n * RNN);
        const int npf = (n + 8 < N) ? n + 8 : n;
        __builtin_prefetch(p_feats + pb + (size_t)npf * RNN + 4 * lane, 0, 0);
        float s = 0.f;
        #pragma unroll
        for (int i = 0; i < 4; ++i) {
            const int a4 = lane + 32 * i;          // float4 index, 128 per row
            const float4 p = prow[a4];
            const float4 hh = hp4[a4];
            const float4 aa = aw4[a4];
            s += tanhf(p.x + hh.x) * aa.x;
            s += tanhf(p.y + hh.y) * aa.y;
            s += tanhf(p.z + hh.z) * aa.z;
            s += tanhf(p.w + hh.w) * aa.w;
        }
        #pragma unroll
        for (int off = 16; off; off >>= 1) s += __shfl_xor(s, off, 32);
        if (lane == 0) sc[n] = s + ab[0];
    }
    __syncthreads();

    // ---- softmax over N (<=196) handled by wave 0
    if (wave == 0) {
        float mx = -3.4e38f;
        for (int n = lane; n < N; n += 32) mx = fmaxf(mx, sc[n]);
        #pragma unroll
        for (int off = 16; off; off >>= 1) mx = fmaxf(mx, __shfl_xor(mx, off, 32));
        float sum = 0.f;
        for (int n = lane; n < N; n += 32) {
            float e = __expf(sc[n] - mx);
            sc[n] = e;
            sum += e;
        }
        #pragma unroll
        for (int off = 16; off; off >>= 1) sum += __shfl_xor(sum, off, 32);
        const float inv = 1.f / sum;
        for (int n = lane; n < N; n += 32) sc[n] *= inv;
    }
    __syncthreads();

    // ---- weighted sum: 2 n-groups x 128 dim-lanes, float4 (b128) loads
    const int dtid = tid & 127;                 // float4 index: dims 4*dtid..+3
    const int grp  = tid >> 7;                  // n parity group
    const float4* frow = (const float4*)(feats + pb);
    float ax = 0.f, ay = 0.f, az = 0.f, awv = 0.f;
    for (int n = grp; n < N; n += 2) {
        const int npf = (n + 4 < N) ? n + 4 : n;
        __builtin_prefetch(frow + (size_t)npf * 128 + dtid, 0, 0);
        const float w = sc[n];
        const float4 f = frow[(size_t)n * 128 + dtid];
        ax += w * f.x;
        ay += w * f.y;
        az += w * f.z;
        awv += w * f.w;
    }
    if (grp == 1) {
        float4 t; t.x = ax; t.y = ay; t.z = az; t.w = awv;
        ((float4*)partial)[dtid] = t;
    }
    __syncthreads();
    if (grp == 0) {
        const float4 p = ((const float4*)partial)[dtid];
        float4 o;
        o.x = ax + p.x; o.y = ay + p.y; o.z = az + p.z; o.w = awv + p.w;
        ((float4*)(outp + (size_t)b * ldout))[dtid] = o;
    }
}

// senti[:, :512] = senti_feats   (float2 per thread)
__global__ __launch_bounds__(256)
void copy_senti_kernel(const float* __restrict__ sf, float* __restrict__ senti)
{
    const int b = blockIdx.x, tid = threadIdx.x;
    ((float2*)(senti + (size_t)b * 1024))[tid] =
        ((const float2*)(sf + (size_t)b * RNN))[tid];
}

// gate = sigmoid(G[b,:] . alpha_w + alpha_b);  out = gate*cont + (1-gate)*senti
__global__ __launch_bounds__(256)
void gate_blend_kernel(const float* __restrict__ G,
                       const float* __restrict__ aw,
                       const float* __restrict__ ab,
                       const float* __restrict__ cont,
                       const float* __restrict__ senti,
                       float* __restrict__ out)
{
    __shared__ float red[256];
    __shared__ float gsh;
    const int b = blockIdx.x, tid = threadIdx.x;

    const float2 gv = ((const float2*)(G + (size_t)b * RNN))[tid];
    const float2 av = ((const float2*)aw)[tid];
    red[tid] = gv.x * av.x + gv.y * av.y;
    __syncthreads();
    for (int st = 128; st > 0; st >>= 1) {
        if (tid < st) red[tid] += red[tid + st];
        __syncthreads();
    }
    if (tid == 0) gsh = 1.f / (1.f + __expf(-(red[0] + ab[0])));
    __syncthreads();

    const float g = gsh;
    const float4 c  = ((const float4*)(cont  + (size_t)b * 1024))[tid];
    const float4 se = ((const float4*)(senti + (size_t)b * 1024))[tid];
    float4 o;
    o.x = g * c.x + (1.f - g) * se.x;
    o.y = g * c.y + (1.f - g) * se.y;
    o.z = g * c.z + (1.f - g) * se.z;
    o.w = g * c.w + (1.f - g) * se.w;
    ((float4*)(out + (size_t)b * 1024))[tid] = o;
}

extern "C" void kernel_launch(void* const* d_in, const int* in_sizes, int n_in,
                              void* d_out, int out_size, void* d_ws, size_t ws_size,
                              hipStream_t stream)
{
    (void)in_sizes; (void)n_in; (void)out_size; (void)ws_size;

    const float* h                  = (const float*)d_in[0];
    const float* att_feats          = (const float*)d_in[1];
    const float* cpt_feats          = (const float*)d_in[2];
    const float* p_att_feats        = (const float*)d_in[3];
    const float* p_cpt_feats        = (const float*)d_in[4];
    const float* senti_feats        = (const float*)d_in[5];
    const float* senti_word_feats   = (const float*)d_in[6];
    const float* p_senti_word_feats = (const float*)d_in[7];
    const float* c_h2att_w = (const float*)d_in[8];
    const float* c_h2att_b = (const float*)d_in[9];
    const float* c_h2cpt_w = (const float*)d_in[10];
    const float* c_h2cpt_b = (const float*)d_in[11];
    const float* c_attA_w  = (const float*)d_in[12];
    const float* c_attA_b  = (const float*)d_in[13];
    const float* c_cptA_w  = (const float*)d_in[14];
    const float* c_cptA_b  = (const float*)d_in[15];
    const float* s_h2word_w = (const float*)d_in[16];
    const float* s_h2word_b = (const float*)d_in[17];
    const float* s_wordA_w  = (const float*)d_in[18];
    const float* s_wordA_b  = (const float*)d_in[19];
    const float* t_h2att_w  = (const float*)d_in[20];
    const float* t_h2att_b  = (const float*)d_in[21];
    const float* t_cont_w   = (const float*)d_in[22];
    const float* t_cont_b   = (const float*)d_in[23];
    const float* t_senti_w  = (const float*)d_in[24];
    const float* t_senti_b  = (const float*)d_in[25];
    const float* t_alpha_w  = (const float*)d_in[26];
    const float* t_alpha_b  = (const float*)d_in[27];

    float* ws    = (float*)d_ws;
    float* HP    = ws;                          // [512, 2048] = h @ [Watt|Wcpt|Wword|Wt] + b
    float* cont  = HP + (size_t)BATCH * 2048;   // [512, 1024] = [att_res | cpt_res]
    float* senti = cont + (size_t)BATCH * 1024; // [512, 1024] = [senti_feats | word_res]
    float* G     = senti + (size_t)BATCH * 1024;// [512, 512]

    const dim3 gg(RNN / 64, BATCH / 16);        // (8, 32)

    // 1) all four h-projections into HP columns [seg*512 .. )
    wmma_gemm_kernel<<<gg, 128, 0, stream>>>(h, RNN, c_h2att_w, RNN, RNN,
        nullptr, 0, nullptr, 0, 0, c_h2att_b, nullptr, nullptr, 0,
        HP + 0, 2048, 0);
    wmma_gemm_kernel<<<gg, 128, 0, stream>>>(h, RNN, c_h2cpt_w, RNN, RNN,
        nullptr, 0, nullptr, 0, 0, c_h2cpt_b, nullptr, nullptr, 0,
        HP + 512, 2048, 0);
    wmma_gemm_kernel<<<gg, 128, 0, stream>>>(h, RNN, s_h2word_w, RNN, RNN,
        nullptr, 0, nullptr, 0, 0, s_h2word_b, nullptr, nullptr, 0,
        HP + 1024, 2048, 0);
    wmma_gemm_kernel<<<gg, 128, 0, stream>>>(h, RNN, t_h2att_w, RNN, RNN,
        nullptr, 0, nullptr, 0, 0, t_h2att_b, nullptr, nullptr, 0,
        HP + 1536, 2048, 0);

    // 2) stage senti_feats into the concatenated senti buffer
    copy_senti_kernel<<<BATCH, 256, 0, stream>>>(senti_feats, senti);

    // 3) the three fused attentions (each feature tensor streamed once)
    attn_kernel<<<BATCH, 256, 0, stream>>>(HP + 0, 2048, p_att_feats, att_feats,
        c_attA_w, c_attA_b, cont + 0, 1024, 196);
    attn_kernel<<<BATCH, 256, 0, stream>>>(HP + 512, 2048, p_cpt_feats, cpt_feats,
        c_cptA_w, c_cptA_b, cont + 512, 1024, 50);
    attn_kernel<<<BATCH, 256, 0, stream>>>(HP + 1024, 2048, p_senti_word_feats,
        senti_word_feats, s_wordA_w, s_wordA_b, senti + 512, 1024, 50);

    // 4) fusion: G = tanh(cont@Wc + senti@Ws + h@Wt + biases)
    wmma_gemm_kernel<<<gg, 128, 0, stream>>>(cont, 1024, t_cont_w, RNN, 1024,
        senti, 1024, t_senti_w, RNN, 1024,
        t_cont_b, t_senti_b, HP + 1536, 2048,
        G, RNN, 1);

    // 5) gate + blend into d_out [512, 1024]
    gate_blend_kernel<<<BATCH, 256, 0, stream>>>(G, t_alpha_w, t_alpha_b,
        cont, senti, (float*)d_out);
}
